// SGD_Hankel_83365315215497
// MI455X (gfx1250) — matrix operand order repaired and checked by reading, not compiled, not run on hardware
//
#include <hip/hip_runtime.h>

typedef __attribute__((ext_vector_type(16))) _Float16 v16h;
typedef __attribute__((ext_vector_type(8)))  float    v8f;

#define NB     65536
#define LL     128
#define DD     4
#define RR     8
#define OO     2
#define NSTEP  128              // 1 first + 126 mid + 1 last
#define WPB    8                // waves per block
#define TPW    2                // independent batch tiles (WMMA chains) per wave
#define TILES  (NB / 16)        // 4096 16-col batch tiles
#define BLOCKS (TILES / (WPB * TPW))   // 256

// ---------------------------------------------------------------------------
// Pack all 128 step matrices (core_first | cores_mid | core_last), TRANSPOSED,
// into a V_WMMA 16x32 f16 A-register layout (contiguous K order), with ROW DUPLICATION:
//   A[M=l][K=kk] = coreT row sel(l),  kk = 4*k + j
//     sel(l) = l      for l in 0..7   (real output ranks)
//     sel(l) = l - 4  for l in 8..11  (duplicate ranks 4..7 so that D rows
//                                      8..11 mirror rows 4..7 -> lanes 16-31
//                                      keep the upper state ranks locally)
//     sel(l) = 0-pad  for l in 12..15
//   lane = l + 16*ha;  element e -> K = e + 16*ha
// Step 0 embeds core_first (K rows k==0 only; initial state is one-hot e0).
// Step 127 embeds core_last (only ranks < O nonzero).
// ---------------------------------------------------------------------------
__global__ __launch_bounds__(256) void tt_pack_cores(
    const float* __restrict__ cf,    // [1][D][R]
    const float* __restrict__ cm,    // [L-2][R][D][R]
    const float* __restrict__ cl,    // [R][D][O]
    _Float16* __restrict__ apack)    // [128][32][16]
{
    int idx = blockIdx.x * blockDim.x + threadIdx.x;   // 0 .. 128*32-1
    if (idx >= NSTEP * 32) return;
    int t    = idx >> 5;
    int lane = idx & 31;
    int l    = lane & 15;            // A-matrix row
    int ha   = lane >> 4;            // which K-interleave this lane holds
    int row  = (l < 8) ? l : (l - 4);   // duplicated rank for rows 8..11
    bool vr  = (l < 12);                // rows 12..15 are zero padding
    _Float16* dst = apack + (size_t)idx * 16;
#pragma unroll
    for (int e = 0; e < 16; ++e) {
        int K = e + 16 * ha;  // contiguous K order, same as build_b
        int k = K >> 2;
        int j = K & 3;
        float v = 0.0f;
        if (vr) {
            if (t == 0) {
                if (k == 0) v = cf[j * RR + row];
            } else if (t <= NSTEP - 2) {
                v = cm[(((size_t)(t - 1) * RR + k) * DD + j) * RR + row];
            } else {
                if (row < OO) v = cl[((size_t)k * DD + j) * OO + row];
            }
        }
        dst[e] = (_Float16)v;
    }
}

// B slice for one chain: element e -> K = 16*h + e, K = 4*k + j; thanks to the
// duplicated A rows, state rank k is this lane's own c[e>>2] for BOTH halves.
__device__ __forceinline__ v16h build_b(const v8f& c, const float4& xv) {
    const float xj[4] = {xv.x, xv.y, xv.z, xv.w};
    v16h b;
#pragma unroll
    for (int j = 0; j < 4; ++j) {
        b[j]      = (_Float16)(c[0] * xj[j]);
        b[4 + j]  = (_Float16)(c[1] * xj[j]);
        b[8 + j]  = (_Float16)(c[2] * xj[j]);
        b[12 + j] = (_Float16)(c[3] * xj[j]);
    }
    return b;
}

// ---------------------------------------------------------------------------
// Main contraction: one wave owns TWO independent 16-col batch tiles, i.e.
// two interleaved WMMA chains. While chain A's WMMA result is in its
// WMMA->VALU hazard window, chain B's VALU/WMMA issue (and vice versa), so
// the 4-coexec hazard nops are filled with useful work. The per-step A tile
// (L2-resident) is loaded once and shared by both chains.
//   D = Apack[t] x B   via v_wmma_f32_16x16x32_f16 (f32 state end to end)
// ---------------------------------------------------------------------------
__global__ __launch_bounds__(256) void tt_wmma_scan(
    const float*    __restrict__ x,      // [N][L][D] f32
    const _Float16* __restrict__ apack,  // [128][32][16] f16
    float*          __restrict__ out)    // [N][O] f32
{
    const int lane = threadIdx.x & 31;
    const int wave = threadIdx.x >> 5;
    const int m    = lane & 15;          // batch column within tile
    const int h    = lane >> 4;          // K-half this lane supplies for B
    const size_t tile = ((size_t)blockIdx.x * WPB + wave) * TPW;
    const size_t n0a  = tile * 16;
    const size_t n0b  = n0a + 16;

    const float* xrowA = x + (n0a + m) * (size_t)(LL * DD);
    const float* xrowB = x + (n0b + m) * (size_t)(LL * DD);

    // initial state = e0 per batch column: D[row 0][col] = 1; duplicated
    // rows 8..11 mirror rows 4..7 which are 0, so upper lanes start at 0.
    v8f ca, cb;
#pragma unroll
    for (int r = 0; r < 8; ++r) { ca[r] = 0.0f; cb[r] = 0.0f; }
    ca[0] = (h == 0) ? 1.0f : 0.0f;
    cb[0] = ca[0];

#pragma unroll 2
    for (int t = 0; t < NSTEP; ++t) {
        // x[n, t, 0:4] — one b128 per lane pair per chain; cacheline reused
        // across 8 consecutive steps
        const float4 xa = *(const float4*)(xrowA + t * DD);
        const float4 xb = *(const float4*)(xrowB + t * DD);

        // A tile for this step (L2-resident, 32 B/lane, shared by both chains)
        const v16h a = *(const v16h*)(apack + ((size_t)t * 32 + lane) * 16);

        const v16h ba = build_b(ca, xa);
        const v16h bb = build_b(cb, xb);

        v8f z;
#pragma unroll
        for (int r = 0; r < 8; ++r) z[r] = 0.0f;
        ca = __builtin_amdgcn_wmma_f32_16x16x32_f16(
                 false, a, false, ba, (short)0, z, false, false);
        cb = __builtin_amdgcn_wmma_f32_16x16x32_f16(
                 false, a, false, bb, (short)0, z, false, false);
    }

    // D rows 0..1 at batch column m hold out[n][0:2] -> contiguous b64 stores
    if (h == 0) {
        float2 oa, ob;
        oa.x = ca[0];
        oa.y = ca[1];
        ob.x = cb[0];
        ob.y = cb[1];
        *(float2*)(out + (n0a + m) * OO) = oa;
        *(float2*)(out + (n0b + m) * OO) = ob;
    }
}

extern "C" void kernel_launch(void* const* d_in, const int* in_sizes, int n_in,
                              void* d_out, int out_size, void* d_ws, size_t ws_size,
                              hipStream_t stream) {
    const float* x  = (const float*)d_in[0];   // [N][L][D]
    const float* cf = (const float*)d_in[1];   // [1][D][R]
    const float* cm = (const float*)d_in[2];   // [L-2][R][D][R]
    const float* cl = (const float*)d_in[3];   // [R][D][O]
    _Float16* apack = (_Float16*)d_ws;         // 128*32*16 f16 = 128 KB
    float* out = (float*)d_out;

    tt_pack_cores<<<(NSTEP * 32 + 255) / 256, 256, 0, stream>>>(cf, cm, cl, apack);
    tt_wmma_scan<<<BLOCKS, 256, 0, stream>>>(x, apack, out);
}
